// DiTBlock_28621662060770
// MI455X (gfx1250) — compile-verified, matrix operand order checked by reading, not run on hardware
//
#include <hip/hip_runtime.h>
#include <hip/hip_bf16.h>
#include <math.h>

// ---------------------------------------------------------------------------
// Problem constants
// ---------------------------------------------------------------------------
constexpr int kB   = 2;
constexpr int kLI  = 1024;
constexpr int kLT  = 256;
constexpr int kL   = kLI + kLT;   // 1280
constexpr int kD   = 768;
constexpr int kH   = 12;
constexpr int kDH  = 64;
constexpr int kE   = 8;
constexpr int kI   = 3072;
constexpr int kIS  = 1536;
constexpr int kMOD = 6 * kD;      // 4608

typedef __attribute__((ext_vector_type(16))) __bf16 v16bf;
typedef __attribute__((ext_vector_type(8)))  float  v8f;

__device__ __forceinline__ v8f wmma_bf16(v16bf a, v16bf b, v8f c) {
  return __builtin_amdgcn_wmma_f32_16x16x32_bf16(false, a, false, b, (short)0, c,
                                                 false, false);
}

__device__ __forceinline__ float siluf(float x) {
  return x / (1.0f + __expf(-x));
}

// ---------------------------------------------------------------------------
// Generic 64x64-tile bf16 WMMA GEMM core.
//   Y[m,n] = sum_k A[m,k] * W[n,k]   (A bf16 row-major lda, W f32 row-major K)
// Block: 256 threads = 8 waves; wave w covers rows (w&3)*16, cols (w>>2)*32.
// Optional row gather for A. Weights are converted f32->bf16 while staging.
// ---------------------------------------------------------------------------
__device__ __forceinline__ void gemm_core(
    const __bf16* __restrict__ A, int lda,
    const int* __restrict__ gather, int M,
    const float* __restrict__ W, int K,
    int m0, int n0, v8f* acc)
{
  __shared__ __bf16 At[64][32];
  __shared__ __bf16 Bt[64][32];

  const int tid   = threadIdx.x;
  const int lane  = tid & 31;
  const int wv    = tid >> 5;
  const int mrow0 = (wv & 3) * 16;
  const int ncol0 = (wv >> 2) * 32;
  const int c15   = lane & 15;
  const int kb8   = (lane >> 4) * 8;

  v8f z = {};
  acc[0] = z; acc[1] = z;

  const int stg_row = tid >> 2;          // 0..63
  const int seg     = (tid & 3) * 8;     // 0,8,16,24

  int arow_g = -1;
  {
    int row = m0 + stg_row;
    if (row < M) arow_g = gather ? gather[row] : row;
  }
  const float* wrow = W + (size_t)(n0 + stg_row) * K + seg;

  for (int k0 = 0; k0 < K; k0 += 32) {
    // ---- stage A tile (bf16 direct copy) ----
    uint4 av = {0u, 0u, 0u, 0u};
    if (arow_g >= 0)
      av = *(const uint4*)(A + (size_t)arow_g * lda + k0 + seg);
    *(uint4*)(&At[stg_row][seg]) = av;

    // ---- stage B tile (f32 -> bf16 convert) ----
    uint4 pk;
    __bf16* tp = (__bf16*)&pk;
#pragma unroll
    for (int j = 0; j < 8; ++j) tp[j] = (__bf16)wrow[k0 + j];
    *(uint4*)(&Bt[stg_row][seg]) = pk;

    if (k0 + 64 < K) __builtin_prefetch(wrow + k0 + 64, 0, 1);

    __syncthreads();

    // ---- fragments (CDNA5 16-bit A/B lane layout) ----
    v16bf af = {}, b0f = {}, b1f = {};
    const __bf16* Ar = &At[mrow0 + c15][0];
    const __bf16* B0 = &Bt[ncol0 + c15][0];
    const __bf16* B1 = &Bt[ncol0 + 16 + c15][0];
#pragma unroll
    for (int j = 0; j < 8; ++j) {
      af[j]      = Ar[kb8 + j];
      af[8 + j]  = Ar[16 + kb8 + j];
      b0f[j]     = B0[kb8 + j];
      b0f[8 + j] = B0[16 + kb8 + j];
      b1f[j]     = B1[kb8 + j];
      b1f[8 + j] = B1[16 + kb8 + j];
    }
    acc[0] = wmma_bf16(af, b0f, acc[0]);
    acc[1] = wmma_bf16(af, b1f, acc[1]);
    __syncthreads();
  }
}

// ---------------------------------------------------------------------------
// GEMM kernels with different epilogues
// ---------------------------------------------------------------------------

// QKV: out = A @ W^T + bias, stored f32 (rows x N)
__global__ __launch_bounds__(256) void k_gemm_bias(
    const __bf16* __restrict__ A, int lda, int M,
    const float* __restrict__ W, int K, int N,
    const float* __restrict__ bias, float* __restrict__ out)
{
  const int m0 = blockIdx.x * 64, n0 = blockIdx.y * 64;
  if (m0 >= M) return;
  v8f acc[2];
  gemm_core(A, lda, nullptr, M, W, K, m0, n0, acc);
  const int lane = threadIdx.x & 31, wv = threadIdx.x >> 5;
  const int mrow0 = (wv & 3) * 16, ncol0 = (wv >> 2) * 32;
  const int c15 = lane & 15, hf = lane >> 4;
#pragma unroll
  for (int nt = 0; nt < 2; ++nt)
#pragma unroll
    for (int r = 0; r < 8; ++r) {
      int row = m0 + mrow0 + r + 8 * hf;
      int col = n0 + ncol0 + nt * 16 + c15;
      if (row < M) out[(size_t)row * N + col] = acc[nt][r] + bias[col];
    }
}

// Projection: out = res + g1[b,col] * (A @ W^T + bias)
__global__ __launch_bounds__(256) void k_gemm_resgate(
    const __bf16* __restrict__ A, int M, int Ltok,
    const float* __restrict__ W, const float* __restrict__ bias,
    const float* __restrict__ res, const float* __restrict__ mod,
    float* __restrict__ out)
{
  const int m0 = blockIdx.x * 64, n0 = blockIdx.y * 64;
  if (m0 >= M) return;
  v8f acc[2];
  gemm_core(A, kD, nullptr, M, W, kD, m0, n0, acc);
  const int lane = threadIdx.x & 31, wv = threadIdx.x >> 5;
  const int mrow0 = (wv & 3) * 16, ncol0 = (wv >> 2) * 32;
  const int c15 = lane & 15, hf = lane >> 4;
#pragma unroll
  for (int nt = 0; nt < 2; ++nt)
#pragma unroll
    for (int r = 0; r < 8; ++r) {
      int row = m0 + mrow0 + r + 8 * hf;
      int col = n0 + ncol0 + nt * 16 + c15;
      if (row < M) {
        int b = row / Ltok;
        float v = acc[nt][r] + bias[col];
        out[(size_t)row * kD + col] =
            res[(size_t)row * kD + col] + mod[b * kMOD + 2 * kD + col] * v;
      }
    }
}

// MoE up: h = silu(A@Wg^T) * (A@Wu^T) -> bf16 (rows x ldh)
__global__ __launch_bounds__(256) void k_gemm_gateup(
    const __bf16* __restrict__ A, const int* __restrict__ lst,
    const int* __restrict__ cnt, int Mmax,
    const float* __restrict__ Wg, const float* __restrict__ Wu,
    int ldh, __bf16* __restrict__ hb)
{
  const int M = cnt ? *cnt : Mmax;
  const int m0 = blockIdx.x * 64, n0 = blockIdx.y * 64;
  if (m0 >= M) return;
  v8f accG[2], accU[2];
  gemm_core(A, kD, lst, M, Wg, kD, m0, n0, accG);
  gemm_core(A, kD, lst, M, Wu, kD, m0, n0, accU);
  const int lane = threadIdx.x & 31, wv = threadIdx.x >> 5;
  const int mrow0 = (wv & 3) * 16, ncol0 = (wv >> 2) * 32;
  const int c15 = lane & 15, hf = lane >> 4;
#pragma unroll
  for (int nt = 0; nt < 2; ++nt)
#pragma unroll
    for (int r = 0; r < 8; ++r) {
      int row = m0 + mrow0 + r + 8 * hf;
      int col = n0 + ncol0 + nt * 16 + c15;
      if (row < M) {
        float g = accG[nt][r], u = accU[nt][r];
        hb[(size_t)row * ldh + col] = (__bf16)(siluf(g) * u);
      }
    }
}

// MoE down: yacc[token] += w * (A@Wd^T), token scattered via list
__global__ __launch_bounds__(256) void k_gemm_down(
    const __bf16* __restrict__ A, int lda,
    const int* __restrict__ cnt, int Mmax,
    const float* __restrict__ Wd, int K,
    const int* __restrict__ lst, const float* __restrict__ wl,
    float* __restrict__ yacc)
{
  const int M = cnt ? *cnt : Mmax;
  const int m0 = blockIdx.x * 64, n0 = blockIdx.y * 64;
  if (m0 >= M) return;
  v8f acc[2];
  gemm_core(A, lda, nullptr, M, Wd, K, m0, n0, acc);
  const int lane = threadIdx.x & 31, wv = threadIdx.x >> 5;
  const int mrow0 = (wv & 3) * 16, ncol0 = (wv >> 2) * 32;
  const int c15 = lane & 15, hf = lane >> 4;
#pragma unroll
  for (int nt = 0; nt < 2; ++nt)
#pragma unroll
    for (int r = 0; r < 8; ++r) {
      int row = m0 + mrow0 + r + 8 * hf;
      int col = n0 + ncol0 + nt * 16 + c15;
      if (row < M) {
        int token = lst ? lst[row] : row;
        float w = wl ? wl[row] : 1.0f;
        yacc[(size_t)token * kD + col] += w * acc[nt][r];
      }
    }
}

// ---------------------------------------------------------------------------
// Modulation: out[set][b][j] = silu(vec[b]) . w[j] + bias[j]   (j in [0,6D))
// ---------------------------------------------------------------------------
__global__ __launch_bounds__(256) void k_mod(
    const float* __restrict__ vec,
    const float* __restrict__ imw, const float* __restrict__ imb,
    const float* __restrict__ tmw, const float* __restrict__ tmb,
    float* __restrict__ modI, float* __restrict__ modT)
{
  int idx = blockIdx.x * 256 + threadIdx.x;          // 2*2*4608 total
  if (idx >= 2 * kB * kMOD) return;
  int j = idx % kMOD;
  int r = idx / kMOD;
  int b = r & 1, set = r >> 1;
  const float* w    = set ? tmw : imw;
  const float* bias = set ? tmb : imb;
  float* out        = set ? modT : modI;
  const float* vp = vec + b * kD;
  float s = 0.f;
  for (int d = 0; d < kD; ++d) s += siluf(vp[d]) * w[(size_t)j * kD + d];
  out[b * kMOD + j] = s + bias[j];
}

// ---------------------------------------------------------------------------
// LN + modulation: out_bf16 = (1 + mod[cIdx]) * LN(x) + mod[sIdx]
// ---------------------------------------------------------------------------
__global__ __launch_bounds__(256) void k_ln_mod(
    const float* __restrict__ x, const float* __restrict__ mod,
    int Ltok, int sIdx, int cIdx, __bf16* __restrict__ out)
{
  __shared__ float red[16];
  __shared__ float stats[2];
  const int token = blockIdx.x;
  const int b = token / Ltok;
  const int tid = threadIdx.x, lane = tid & 31, wv = tid >> 5;
  const float* xr = x + (size_t)token * kD;

  float s = 0.f, s2 = 0.f;
  for (int j = tid; j < kD; j += 256) { float v = xr[j]; s += v; s2 += v * v; }
#pragma unroll
  for (int o = 16; o; o >>= 1) { s += __shfl_xor(s, o, 32); s2 += __shfl_xor(s2, o, 32); }
  if (lane == 0) { red[wv] = s; red[8 + wv] = s2; }
  __syncthreads();
  if (tid == 0) {
    float a = 0.f, q = 0.f;
    for (int i = 0; i < 8; ++i) { a += red[i]; q += red[8 + i]; }
    float mean = a / kD;
    float var  = q / kD - mean * mean;
    stats[0] = mean;
    stats[1] = rsqrtf(var + 1e-6f);
  }
  __syncthreads();
  const float mean = stats[0], rstd = stats[1];
  const float* sm = mod + b * kMOD + sIdx * kD;
  const float* cm = mod + b * kMOD + cIdx * kD;
  for (int j = tid; j < kD; j += 256) {
    float xn = (xr[j] - mean) * rstd;
    out[(size_t)token * kD + j] = (__bf16)((1.f + cm[j]) * xn + sm[j]);
  }
}

// ---------------------------------------------------------------------------
// RMS-norm q/k + scatter qkv (f32 rows of 2304) -> q/k/v bf16 [B,H,L,DH]
// ---------------------------------------------------------------------------
__global__ __launch_bounds__(256) void k_rms_scatter(
    const float* __restrict__ qkv,
    const float* __restrict__ qs, const float* __restrict__ ks,
    __bf16* __restrict__ qb, __bf16* __restrict__ kb, __bf16* __restrict__ vb,
    int Ltok, int lOff)
{
  const int token = blockIdx.x;
  const int b = token / Ltok;
  const int pos = token % Ltok;
  const int l = lOff + pos;
  const int tid = threadIdx.x, lane = tid & 31, wv = tid >> 5;

  for (int g = wv; g < 3 * kH; g += 8) {
    int comp = g / kH, h = g % kH;
    size_t base = (size_t)token * (3 * kD) + (size_t)(comp * kH + h) * kDH;
    float x0 = qkv[base + lane];
    float x1 = qkv[base + 32 + lane];
    float y0 = x0, y1 = x1;
    if (comp < 2) {
      float ss = x0 * x0 + x1 * x1;
#pragma unroll
      for (int o = 16; o; o >>= 1) ss += __shfl_xor(ss, o, 32);
      float r = rsqrtf(ss / kDH + 1e-6f);
      const float* sc = (comp == 0) ? qs : ks;
      y0 = x0 * r * sc[lane];
      y1 = x1 * r * sc[32 + lane];
    }
    __bf16* dst = (comp == 0) ? qb : (comp == 1) ? kb : vb;
    size_t drow = ((size_t)(b * kH + h) * kL + l) * kDH;
    dst[drow + lane]      = (__bf16)y0;
    dst[drow + 32 + lane] = (__bf16)y1;
  }
}

// ---------------------------------------------------------------------------
// Attention: one wave per (b, h, 16-row q tile). WMMA QK^T + softmax + WMMA PV.
// Scores kept as bf16 in LDS (16 x 1280).
// ---------------------------------------------------------------------------
__global__ __launch_bounds__(32) void k_attn(
    const __bf16* __restrict__ qg, const __bf16* __restrict__ kg,
    const __bf16* __restrict__ vg,
    __bf16* __restrict__ aT, __bf16* __restrict__ aI)
{
  __shared__ __bf16 sc[16][kL];      // 40 KB
  __shared__ __bf16 qt[16 * 64];     // 2 KB
  __shared__ __bf16 kv[32 * 64];     // 4 KB
  __shared__ float  mx[16], isum[16];

  const int lane = threadIdx.x;
  const int qt0 = blockIdx.x * 16;
  const int h = blockIdx.y;
  const int b = blockIdx.z;
  const size_t base = (size_t)(b * kH + h) * kL * kDH;

  const int c15 = lane & 15;
  const int kb8 = (lane >> 4) * 8;

  // load q tile (loop copies 256 uint4 = 32*64 bf16; qt holds 16*64)
  {
    const uint4* src = (const uint4*)(qg + base + (size_t)qt0 * kDH);
    uint4* dst = (uint4*)qt;
    for (int i = lane; i < 256; i += 32) dst[i] = src[i];
  }
  __syncthreads();

  v16bf a0 = {}, a1 = {};
  {
    const __bf16* Ar = qt + c15 * 64;
#pragma unroll
    for (int j = 0; j < 8; ++j) {
      a0[j]     = Ar[kb8 + j];      a0[8 + j] = Ar[16 + kb8 + j];
      a1[j]     = Ar[32 + kb8 + j]; a1[8 + j] = Ar[48 + kb8 + j];
    }
  }

  // ---- QK^T ----
  for (int kt = 0; kt < kL / 16; ++kt) {
    {
      const uint4* src = (const uint4*)(kg + base + (size_t)kt * 16 * kDH);
      uint4* dst = (uint4*)kv;
      for (int i = lane; i < 128; i += 32) dst[i] = src[i];
    }
    __syncthreads();
    v16bf b0 = {}, b1 = {};
    const __bf16* Br = kv + c15 * 64;
#pragma unroll
    for (int j = 0; j < 8; ++j) {
      b0[j]     = Br[kb8 + j];      b0[8 + j] = Br[16 + kb8 + j];
      b1[j]     = Br[32 + kb8 + j]; b1[8 + j] = Br[48 + kb8 + j];
    }
    v8f c = {};
    c = wmma_bf16(a0, b0, c);
    c = wmma_bf16(a1, b1, c);
    const int hf = lane >> 4;
#pragma unroll
    for (int r = 0; r < 8; ++r)
      sc[r + 8 * hf][kt * 16 + c15] = (__bf16)(c[r] * 0.125f);
    __syncthreads();
  }

  // ---- softmax stats ----
  for (int r = 0; r < 16; ++r) {
    float m = -1e30f;
    for (int j = lane; j < kL; j += 32) m = fmaxf(m, (float)sc[r][j]);
#pragma unroll
    for (int o = 16; o; o >>= 1) m = fmaxf(m, __shfl_xor(m, o, 32));
    float s = 0.f;
    for (int j = lane; j < kL; j += 32) s += __expf((float)sc[r][j] - m);
#pragma unroll
    for (int o = 16; o; o >>= 1) s += __shfl_xor(s, o, 32);
    if (lane == 0) { mx[r] = m; isum[r] = 1.0f / s; }
  }
  __syncthreads();

  // ---- P @ V ----
  v8f acc[4];
  v8f z = {};
#pragma unroll
  for (int nt = 0; nt < 4; ++nt) acc[nt] = z;

  const float mxr = mx[c15], isr = isum[c15];
  for (int kt = 0; kt < kL / 32; ++kt) {
    const int kk0 = kt * 32;
    {
      const uint4* src = (const uint4*)(vg + base + (size_t)kk0 * kDH);
      uint4* dst = (uint4*)kv;
      for (int i = lane; i < 256; i += 32) dst[i] = src[i];
    }
    __syncthreads();
    v16bf ap = {};
#pragma unroll
    for (int j = 0; j < 8; ++j) {
      ap[j]     = (__bf16)(__expf((float)sc[c15][kk0 + kb8 + j] - mxr) * isr);
      ap[8 + j] = (__bf16)(__expf((float)sc[c15][kk0 + 16 + kb8 + j] - mxr) * isr);
    }
#pragma unroll
    for (int nt = 0; nt < 4; ++nt) {
      v16bf bf = {};
      const int col = nt * 16 + c15;
#pragma unroll
      for (int j = 0; j < 8; ++j) {
        bf[j]     = kv[(kb8 + j) * 64 + col];
        bf[8 + j] = kv[(16 + kb8 + j) * 64 + col];
      }
      acc[nt] = wmma_bf16(ap, bf, acc[nt]);
    }
    __syncthreads();
  }

  // ---- store (split txt/img token streams) ----
  const int hf = lane >> 4;
#pragma unroll
  for (int nt = 0; nt < 4; ++nt)
#pragma unroll
    for (int r = 0; r < 8; ++r) {
      int m = r + 8 * hf;
      int l = qt0 + m;
      int col = h * kDH + nt * 16 + c15;
      __bf16 v = (__bf16)acc[nt][r];
      if (l < kLT) aT[((size_t)(b * kLT + l)) * kD + col] = v;
      else         aI[((size_t)(b * kLI + (l - kLT))) * kD + col] = v;
    }
}

// ---------------------------------------------------------------------------
// Router: softmax over 8 expert logits, top-2, push into gather lists
// ---------------------------------------------------------------------------
__global__ __launch_bounds__(256) void k_router(
    const __bf16* __restrict__ x2, const float* __restrict__ gw,
    int* __restrict__ cnt, int* __restrict__ lst, float* __restrict__ wl)
{
  __shared__ float scs[kE];
  const int token = blockIdx.x;
  const int tid = threadIdx.x, lane = tid & 31, wv = tid >> 5;
  if (wv < kE) {
    float s = 0.f;
    const __bf16* xr = x2 + (size_t)token * kD;
    const float* g = gw + (size_t)wv * kD;
    for (int j = lane; j < kD; j += 32) s += (float)xr[j] * g[j];
#pragma unroll
    for (int o = 16; o; o >>= 1) s += __shfl_xor(s, o, 32);
    if (lane == 0) scs[wv] = s;
  }
  __syncthreads();
  if (tid == 0) {
    float m = scs[0];
    for (int i = 1; i < kE; ++i) m = fmaxf(m, scs[i]);
    float p[kE], sum = 0.f;
    for (int i = 0; i < kE; ++i) { p[i] = __expf(scs[i] - m); sum += p[i]; }
    float inv = 1.0f / sum;
    for (int i = 0; i < kE; ++i) p[i] *= inv;
    int i0 = 0;
    for (int i = 1; i < kE; ++i) if (p[i] > p[i0]) i0 = i;
    int i1 = (i0 == 0) ? 1 : 0;
    for (int i = 0; i < kE; ++i) if (i != i0 && p[i] > p[i1]) i1 = i;
    int s0 = atomicAdd(&cnt[i0], 1);
    lst[i0 * 2048 + s0] = token; wl[i0 * 2048 + s0] = p[i0];
    int s1 = atomicAdd(&cnt[i1], 1);
    lst[i1 * 2048 + s1] = token; wl[i1 * 2048 + s1] = p[i1];
  }
}

// ---------------------------------------------------------------------------
// Final gated residual add: out += g2[b,j] * yacc
// ---------------------------------------------------------------------------
__global__ __launch_bounds__(256) void k_final_add(
    float* __restrict__ out, const float* __restrict__ yacc,
    const float* __restrict__ mod, int Ltok, int total)
{
  int i = blockIdx.x * 256 + threadIdx.x;
  if (i >= total) return;
  int j = i % kD;
  int token = i / kD;
  int b = token / Ltok;
  out[i] += mod[b * kMOD + 5 * kD + j] * yacc[i];
}

// ---------------------------------------------------------------------------
// Host-side orchestration
// ---------------------------------------------------------------------------
extern "C" void kernel_launch(void* const* d_in, const int* in_sizes, int n_in,
                              void* d_out, int out_size, void* d_ws, size_t ws_size,
                              hipStream_t stream)
{
  const float* img   = (const float*)d_in[0];
  const float* txt   = (const float*)d_in[1];
  const float* vec   = (const float*)d_in[2];
  const float* imw   = (const float*)d_in[3];
  const float* imb   = (const float*)d_in[4];
  const float* tmw   = (const float*)d_in[5];
  const float* tmb   = (const float*)d_in[6];
  const float* iqkvw = (const float*)d_in[7];
  const float* iqkvb = (const float*)d_in[8];
  const float* ipw   = (const float*)d_in[9];
  const float* ipb   = (const float*)d_in[10];
  const float* tqkvw = (const float*)d_in[11];
  const float* tqkvb = (const float*)d_in[12];
  const float* tpw   = (const float*)d_in[13];
  const float* tpb   = (const float*)d_in[14];
  const float* iqs   = (const float*)d_in[15];
  const float* iks   = (const float*)d_in[16];
  const float* tqs   = (const float*)d_in[17];
  const float* tks   = (const float*)d_in[18];
  const float* igw   = (const float*)d_in[19];
  const float* ieg   = (const float*)d_in[20];
  const float* ieu   = (const float*)d_in[21];
  const float* ied   = (const float*)d_in[22];
  const float* isg   = (const float*)d_in[23];
  const float* isu   = (const float*)d_in[24];
  const float* isd   = (const float*)d_in[25];
  const float* tgw   = (const float*)d_in[26];
  const float* teg   = (const float*)d_in[27];
  const float* teu   = (const float*)d_in[28];
  const float* ted   = (const float*)d_in[29];
  const float* tsg   = (const float*)d_in[30];
  const float* tsu   = (const float*)d_in[31];
  const float* tsd   = (const float*)d_in[32];

  const int TI = kB * kLI;   // 2048 img tokens
  const int TT = kB * kLT;   // 512 txt tokens

  char* ws = (char*)d_ws;
  size_t off = 0;
  auto take = [&](size_t bytes) -> char* {
    char* p = ws + off;
    off = (off + bytes + 255) & ~(size_t)255;
    return p;
  };

  float*  modI  = (float*) take((size_t)kB * kMOD * 4);
  float*  modT  = (float*) take((size_t)kB * kMOD * 4);
  __bf16* xmodI = (__bf16*)take((size_t)TI * kD * 2);
  __bf16* xmodT = (__bf16*)take((size_t)TT * kD * 2);
  float*  qkvI  = (float*) take((size_t)TI * 3 * kD * 4);
  float*  qkvT  = (float*) take((size_t)TT * 3 * kD * 4);
  __bf16* qb    = (__bf16*)take((size_t)kB * kH * kL * kDH * 2);
  __bf16* kb    = (__bf16*)take((size_t)kB * kH * kL * kDH * 2);
  __bf16* vb    = (__bf16*)take((size_t)kB * kH * kL * kDH * 2);
  __bf16* attnI = (__bf16*)take((size_t)TI * kD * 2);
  __bf16* attnT = (__bf16*)take((size_t)TT * kD * 2);
  __bf16* x2I   = (__bf16*)take((size_t)TI * kD * 2);
  __bf16* x2T   = (__bf16*)take((size_t)TT * kD * 2);
  float*  yaccI = (float*) take((size_t)TI * kD * 4);
  float*  yaccT = (float*) take((size_t)TT * kD * 4);
  __bf16* hbuf  = (__bf16*)take((size_t)TI * kI * 2);
  int*    cntI  = (int*)   take(kE * 4);
  int*    cntT  = (int*)   take(kE * 4);
  int*    lstI  = (int*)   take((size_t)kE * 2048 * 4);
  int*    lstT  = (int*)   take((size_t)kE * 2048 * 4);
  float*  wlI   = (float*) take((size_t)kE * 2048 * 4);
  float*  wlT   = (float*) take((size_t)kE * 2048 * 4);

  float* outImg = (float*)d_out;
  float* outTxt = (float*)d_out + (size_t)TI * kD;

  // zero accumulators / counters (graph-capture legal)
  hipMemsetAsync(cntI, 0, kE * 4, stream);
  hipMemsetAsync(cntT, 0, kE * 4, stream);
  hipMemsetAsync(yaccI, 0, (size_t)TI * kD * 4, stream);
  hipMemsetAsync(yaccT, 0, (size_t)TT * kD * 4, stream);

  // 1. modulation vectors
  k_mod<<<(2 * kB * kMOD + 255) / 256, 256, 0, stream>>>(
      vec, imw, imb, tmw, tmb, modI, modT);

  // 2. LN + (1+c1)*x + s1 -> bf16
  k_ln_mod<<<TI, 256, 0, stream>>>(img, modI, kLI, 0, 1, xmodI);
  k_ln_mod<<<TT, 256, 0, stream>>>(txt, modT, kLT, 0, 1, xmodT);

  // 3. QKV GEMMs (f32 out with bias)
  k_gemm_bias<<<dim3(TI / 64, (3 * kD) / 64), 256, 0, stream>>>(
      xmodI, kD, TI, iqkvw, kD, 3 * kD, iqkvb, qkvI);
  k_gemm_bias<<<dim3(TT / 64, (3 * kD) / 64), 256, 0, stream>>>(
      xmodT, kD, TT, tqkvw, kD, 3 * kD, tqkvb, qkvT);

  // 4. RMS norm q/k + scatter to [B,H,L,DH] bf16 (txt at l=0..255, img after)
  k_rms_scatter<<<TI, 256, 0, stream>>>(qkvI, iqs, iks, qb, kb, vb, kLI, kLT);
  k_rms_scatter<<<TT, 256, 0, stream>>>(qkvT, tqs, tks, qb, kb, vb, kLT, 0);

  // 5. joint attention
  k_attn<<<dim3(kL / 16, kH, kB), 32, 0, stream>>>(qb, kb, vb, attnT, attnI);

  // 6. output projection + gated residual -> d_out
  k_gemm_resgate<<<dim3(TI / 64, kD / 64), 256, 0, stream>>>(
      attnI, TI, kLI, ipw, ipb, img, modI, outImg);
  k_gemm_resgate<<<dim3(TT / 64, kD / 64), 256, 0, stream>>>(
      attnT, TT, kLT, tpw, tpb, txt, modT, outTxt);

  // 7. MoE pre-norm (reads residual in d_out): (1+c2)*LN + s2 -> bf16
  k_ln_mod<<<TI, 256, 0, stream>>>(outImg, modI, kLI, 3, 4, x2I);
  k_ln_mod<<<TT, 256, 0, stream>>>(outTxt, modT, kLT, 3, 4, x2T);

  // 8. routing (top-2 of softmax)
  k_router<<<TI, 256, 0, stream>>>(x2I, igw, cntI, lstI, wlI);
  k_router<<<TT, 256, 0, stream>>>(x2T, tgw, cntT, lstT, wlT);

  // 9. routed experts, img stream
  for (int e = 0; e < kE; ++e) {
    k_gemm_gateup<<<dim3(TI / 64, kI / 64), 256, 0, stream>>>(
        x2I, lstI + e * 2048, cntI + e, TI,
        ieg + (size_t)e * kI * kD, ieu + (size_t)e * kI * kD, kI, hbuf);
    k_gemm_down<<<dim3(TI / 64, kD / 64), 256, 0, stream>>>(
        hbuf, kI, cntI + e, TI, ied + (size_t)e * kD * kI, kI,
        lstI + e * 2048, wlI + e * 2048, yaccI);
  }
  // shared expert, img
  k_gemm_gateup<<<dim3(TI / 64, kIS / 64), 256, 0, stream>>>(
      x2I, nullptr, nullptr, TI, isg, isu, kIS, hbuf);
  k_gemm_down<<<dim3(TI / 64, kD / 64), 256, 0, stream>>>(
      hbuf, kIS, nullptr, TI, isd, kIS, nullptr, nullptr, yaccI);

  // 10. routed experts, txt stream
  for (int e = 0; e < kE; ++e) {
    k_gemm_gateup<<<dim3(TT / 64, kI / 64), 256, 0, stream>>>(
        x2T, lstT + e * 2048, cntT + e, TT,
        teg + (size_t)e * kI * kD, teu + (size_t)e * kI * kD, kI, hbuf);
    k_gemm_down<<<dim3(TT / 64, kD / 64), 256, 0, stream>>>(
        hbuf, kI, cntT + e, TT, ted + (size_t)e * kD * kI, kI,
        lstT + e * 2048, wlT + e * 2048, yaccT);
  }
  // shared expert, txt
  k_gemm_gateup<<<dim3(TT / 64, kIS / 64), 256, 0, stream>>>(
      x2T, nullptr, nullptr, TT, tsg, tsu, kIS, hbuf);
  k_gemm_down<<<dim3(TT / 64, kD / 64), 256, 0, stream>>>(
      hbuf, kIS, nullptr, TT, tsd, kIS, nullptr, nullptr, yaccT);

  // 11. final gated residual adds
  k_final_add<<<(TI * kD + 255) / 256, 256, 0, stream>>>(
      outImg, yaccI, modI, kLI, TI * kD);
  k_final_add<<<(TT * kD + 255) / 256, 256, 0, stream>>>(
      outTxt, yaccT, modT, kLT, TT * kD);
}
